// CausalSelfAttention_44203803410775
// MI455X (gfx1250) — matrix operand order repaired and checked by reading, not compiled, not run on hardware
//
#include <hip/hip_runtime.h>
#include <hip/hip_bf16.h>

typedef __bf16 bf16;
typedef __attribute__((ext_vector_type(16))) __bf16 v16bf;
typedef __attribute__((ext_vector_type(8)))  float  v8f;
typedef __attribute__((ext_vector_type(4)))  unsigned int u32x4;

#define DIM   768
#define S_LEN 4096
#define NH    12
#define DH    64

union FragBF { v16bf v; u32x4 q[2]; };

static __device__ __forceinline__ v8f wmma_bf16(const FragBF& a, const FragBF& b, v8f c) {
  return __builtin_amdgcn_wmma_f32_16x16x32_bf16(false, a.v, false, b.v, (short)0, c, false, false);
}

// xor-shuffle within 16-lane half-groups via ds_swizzle (group-of-32 mode, and=0x1f)
template<int XM>
static __device__ __forceinline__ float swz_xor(float v) {
  int i = __builtin_amdgcn_ds_swizzle(__builtin_bit_cast(int, v), (XM << 10) | 0x1f);
  return __builtin_bit_cast(float, i);
}

// ---------------------------------------------------------------- converts
__global__ __launch_bounds__(256) void convert_x_kernel(const float* __restrict__ x,
                                                        bf16* __restrict__ xb, int n) {
  int i = blockIdx.x * 256 + threadIdx.x;
  if (i < n) xb[i] = (bf16)x[i];
}

// Wt[n][k] = (bf16) W[k][n]   (768x768), tiled for coalescing
__global__ __launch_bounds__(256) void transpose_convert_kernel(const float* __restrict__ W,
                                                                bf16* __restrict__ Wt) {
  __shared__ float tile[32][33];
  const int tx = threadIdx.x, ty = threadIdx.y;       // blockDim = (32,8)
  const int bn = blockIdx.x * 32, bk = blockIdx.y * 32;
#pragma unroll
  for (int j = 0; j < 32; j += 8)
    tile[ty + j][tx] = W[(bk + ty + j) * DIM + bn + tx];
  __syncthreads();
#pragma unroll
  for (int j = 0; j < 32; j += 8)
    Wt[(bn + ty + j) * DIM + bk + tx] = (bf16)tile[tx][ty + j];
}

// ---------------------------------------------------------------- QKV + RoPE
// grid (64, NH, 3), block 128 (4 waves). Each wave: 16x64 stripe of one head.
// sel 0 -> q [h][S][64] (RoPE), 1 -> k [h][S][64] (RoPE), 2 -> v transposed [h][64][S]
__global__ __launch_bounds__(128) void qkv_rope_kernel(
    const bf16* __restrict__ xb,
    const bf16* __restrict__ Wqt, const bf16* __restrict__ Wkt, const bf16* __restrict__ Wvt,
    const int* __restrict__ posids,
    bf16* __restrict__ qh, bf16* __restrict__ kh, bf16* __restrict__ vt) {
  const int lane = threadIdx.x & 31;
  const int wave = threadIdx.x >> 5;
  const int ln = lane & 15, half = lane >> 4;
  const int m0 = (blockIdx.x * 4 + wave) * 16;
  const int h = blockIdx.y, sel = blockIdx.z;
  const bf16* Wt = (sel == 0) ? Wqt : (sel == 1) ? Wkt : Wvt;

  const bf16* arow  = xb + (m0 + ln) * DIM;
  const bf16* brow0 = Wt + (h * DH +  0 + ln) * DIM;
  const bf16* brow1 = Wt + (h * DH + 16 + ln) * DIM;
  const bf16* brow2 = Wt + (h * DH + 32 + ln) * DIM;
  const bf16* brow3 = Wt + (h * DH + 48 + ln) * DIM;

  v8f acc0 = {}, acc1 = {}, acc2 = {}, acc3 = {};
  for (int k0 = 0; k0 < DIM; k0 += 32) {
    FragBF a, b0, b1, b2, b3;
    a.q[0] = *(const u32x4*)(arow + k0 + 16 * half);
    a.q[1] = *(const u32x4*)(arow + k0 + 8 + 16 * half);
    const int bo = k0 + 16 * half;
    b0.q[0] = *(const u32x4*)(brow0 + bo); b0.q[1] = *(const u32x4*)(brow0 + bo + 8);
    b1.q[0] = *(const u32x4*)(brow1 + bo); b1.q[1] = *(const u32x4*)(brow1 + bo + 8);
    b2.q[0] = *(const u32x4*)(brow2 + bo); b2.q[1] = *(const u32x4*)(brow2 + bo + 8);
    b3.q[0] = *(const u32x4*)(brow3 + bo); b3.q[1] = *(const u32x4*)(brow3 + bo + 8);
    acc0 = wmma_bf16(a, b0, acc0);
    acc1 = wmma_bf16(a, b1, acc1);
    acc2 = wmma_bf16(a, b2, acc2);
    acc3 = wmma_bf16(a, b3, acc3);
  }

  if (sel < 2) {
    // RoPE: cols of head: acc0 -> j=ln, acc1 -> 16+ln, acc2 -> 32+ln, acc3 -> 48+ln
    const float C = 0.28782313662425845f;  // ln(10000)/32
    const float invA = __expf(-(float)ln * C);
    const float invB = __expf(-(float)(ln + 16) * C);
    bf16* outp = (sel == 0) ? qh : kh;
#pragma unroll
    for (int r = 0; r < 8; ++r) {
      const int mm = m0 + r + 8 * half;
      const float p = (float)posids[mm];
      const float aA = p * invA, aB = p * invB;
      const float sA = __sinf(aA), cA = __cosf(aA);
      const float sB = __sinf(aB), cB = __cosf(aB);
      const float o0 = acc0[r] * cA - acc2[r] * sA;
      const float o2 = acc2[r] * cA + acc0[r] * sA;
      const float o1 = acc1[r] * cB - acc3[r] * sB;
      const float o3 = acc3[r] * cB + acc1[r] * sB;
      bf16* row = outp + (h * S_LEN + mm) * DH;
      row[ln] = (bf16)o0; row[16 + ln] = (bf16)o1;
      row[32 + ln] = (bf16)o2; row[48 + ln] = (bf16)o3;
    }
  } else {
#pragma unroll
    for (int r = 0; r < 8; ++r) {
      const int mm = m0 + r + 8 * half;
      vt[(h * DH +  0 + ln) * S_LEN + mm] = (bf16)acc0[r];
      vt[(h * DH + 16 + ln) * S_LEN + mm] = (bf16)acc1[r];
      vt[(h * DH + 32 + ln) * S_LEN + mm] = (bf16)acc2[r];
      vt[(h * DH + 48 + ln) * S_LEN + mm] = (bf16)acc3[r];
    }
  }
}

// ---------------------------------------------------------------- flash attention
// grid (S/16, NH), block 32 (one wave). Online softmax, 32-key blocks.
__global__ __launch_bounds__(32) void attn_kernel(
    const bf16* __restrict__ qh, const bf16* __restrict__ kh,
    const bf16* __restrict__ vt, bf16* __restrict__ attnb) {
  __shared__ __align__(16) bf16 pb[16 * 32];
  const int lane = threadIdx.x & 31;
  const int ln = lane & 15, half = lane >> 4;
  const int qb = blockIdx.x, h = blockIdx.y;
  const int m0 = qb * 16;

  const bf16* qrow = qh + (h * S_LEN + m0 + ln) * DH;
  FragBF qa0, qa1;
  qa0.q[0] = *(const u32x4*)(qrow + 16 * half);
  qa0.q[1] = *(const u32x4*)(qrow + 8 + 16 * half);
  qa1.q[0] = *(const u32x4*)(qrow + 32 + 16 * half);
  qa1.q[1] = *(const u32x4*)(qrow + 40 + 16 * half);

  v8f o0 = {}, o1 = {}, o2 = {}, o3 = {};
  float mrow[8], lrow[8];
#pragma unroll
  for (int r = 0; r < 8; ++r) { mrow[r] = -3.0e38f; lrow[r] = 0.f; }

  const int nkb = (m0 + 16 + 31) >> 5;
  for (int kb = 0; kb < nkb; ++kb) {
    const int kbase = kb * 32;
    v8f s0 = {}, s1 = {};
    {
      const bf16* kr0 = kh + (h * S_LEN + kbase + ln) * DH;
      const bf16* kr1 = kh + (h * S_LEN + kbase + 16 + ln) * DH;
      FragBF b;
      b.q[0] = *(const u32x4*)(kr0 + 16 * half);      b.q[1] = *(const u32x4*)(kr0 + 8 + 16 * half);
      s0 = wmma_bf16(qa0, b, s0);
      b.q[0] = *(const u32x4*)(kr0 + 32 + 16 * half); b.q[1] = *(const u32x4*)(kr0 + 40 + 16 * half);
      s0 = wmma_bf16(qa1, b, s0);
      b.q[0] = *(const u32x4*)(kr1 + 16 * half);      b.q[1] = *(const u32x4*)(kr1 + 8 + 16 * half);
      s1 = wmma_bf16(qa0, b, s1);
      b.q[0] = *(const u32x4*)(kr1 + 32 + 16 * half); b.q[1] = *(const u32x4*)(kr1 + 40 + 16 * half);
      s1 = wmma_bf16(qa1, b, s1);
    }
    const bool needMask = (kbase + 31) > m0;
#pragma unroll
    for (int r = 0; r < 8; ++r) {
      const int row = m0 + r + 8 * half;
      float v0 = s0[r] * 0.125f;
      float v1 = s1[r] * 0.125f;
      if (needMask) {
        if (kbase + ln > row)      v0 = -3.0e38f;
        if (kbase + 16 + ln > row) v1 = -3.0e38f;
      }
      float mx = fmaxf(v0, v1);
      mx = fmaxf(mx, swz_xor<1>(mx));
      mx = fmaxf(mx, swz_xor<2>(mx));
      mx = fmaxf(mx, swz_xor<4>(mx));
      mx = fmaxf(mx, swz_xor<8>(mx));
      const float mnew = fmaxf(mrow[r], mx);
      const float f = __expf(mrow[r] - mnew);
      const float p0 = __expf(v0 - mnew);
      const float p1 = __expf(v1 - mnew);
      float rs = p0 + p1;
      rs += swz_xor<1>(rs); rs += swz_xor<2>(rs);
      rs += swz_xor<4>(rs); rs += swz_xor<8>(rs);
      lrow[r] = lrow[r] * f + rs;
      mrow[r] = mnew;
      o0[r] *= f; o1[r] *= f; o2[r] *= f; o3[r] *= f;
      pb[(r + 8 * half) * 32 + ln]      = (bf16)p0;
      pb[(r + 8 * half) * 32 + 16 + ln] = (bf16)p1;
    }
    __syncthreads();                 // single-wave WG: S_NOP + DS fence
    FragBF pf;
    const bf16* prow = pb + ln * 32;
    pf.q[0] = *(const u32x4*)(prow + 16 * half);
    pf.q[1] = *(const u32x4*)(prow + 8 + 16 * half);
    __syncthreads();
    {
      FragBF vb;
      const bf16* vr;
      vr = vt + (h * DH +  0 + ln) * S_LEN + kbase + 16 * half;
      vb.q[0] = *(const u32x4*)vr; vb.q[1] = *(const u32x4*)(vr + 8);
      o0 = wmma_bf16(pf, vb, o0);
      vr = vt + (h * DH + 16 + ln) * S_LEN + kbase + 16 * half;
      vb.q[0] = *(const u32x4*)vr; vb.q[1] = *(const u32x4*)(vr + 8);
      o1 = wmma_bf16(pf, vb, o1);
      vr = vt + (h * DH + 32 + ln) * S_LEN + kbase + 16 * half;
      vb.q[0] = *(const u32x4*)vr; vb.q[1] = *(const u32x4*)(vr + 8);
      o2 = wmma_bf16(pf, vb, o2);
      vr = vt + (h * DH + 48 + ln) * S_LEN + kbase + 16 * half;
      vb.q[0] = *(const u32x4*)vr; vb.q[1] = *(const u32x4*)(vr + 8);
      o3 = wmma_bf16(pf, vb, o3);
    }
  }
#pragma unroll
  for (int r = 0; r < 8; ++r) {
    const float inv = 1.0f / lrow[r];
    bf16* orow = attnb + (m0 + r + 8 * half) * DIM + h * DH;
    orow[ln]      = (bf16)(o0[r] * inv);
    orow[16 + ln] = (bf16)(o1[r] * inv);
    orow[32 + ln] = (bf16)(o2[r] * inv);
    orow[48 + ln] = (bf16)(o3[r] * inv);
  }
}

// ---------------------------------------------------------------- out projection
__global__ __launch_bounds__(128) void oproj_kernel(const bf16* __restrict__ attnb,
                                                    const bf16* __restrict__ Wot,
                                                    float* __restrict__ out) {
  const int lane = threadIdx.x & 31;
  const int wave = threadIdx.x >> 5;
  const int ln = lane & 15, half = lane >> 4;
  const int m0 = (blockIdx.x * 4 + wave) * 16;
  const int n0 = blockIdx.y * 64;

  const bf16* arow  = attnb + (m0 + ln) * DIM;
  const bf16* brow0 = Wot + (n0 +  0 + ln) * DIM;
  const bf16* brow1 = Wot + (n0 + 16 + ln) * DIM;
  const bf16* brow2 = Wot + (n0 + 32 + ln) * DIM;
  const bf16* brow3 = Wot + (n0 + 48 + ln) * DIM;

  v8f acc0 = {}, acc1 = {}, acc2 = {}, acc3 = {};
  for (int k0 = 0; k0 < DIM; k0 += 32) {
    FragBF a, b0, b1, b2, b3;
    a.q[0] = *(const u32x4*)(arow + k0 + 16 * half);
    a.q[1] = *(const u32x4*)(arow + k0 + 8 + 16 * half);
    const int bo = k0 + 16 * half;
    b0.q[0] = *(const u32x4*)(brow0 + bo); b0.q[1] = *(const u32x4*)(brow0 + bo + 8);
    b1.q[0] = *(const u32x4*)(brow1 + bo); b1.q[1] = *(const u32x4*)(brow1 + bo + 8);
    b2.q[0] = *(const u32x4*)(brow2 + bo); b2.q[1] = *(const u32x4*)(brow2 + bo + 8);
    b3.q[0] = *(const u32x4*)(brow3 + bo); b3.q[1] = *(const u32x4*)(brow3 + bo + 8);
    acc0 = wmma_bf16(a, b0, acc0);
    acc1 = wmma_bf16(a, b1, acc1);
    acc2 = wmma_bf16(a, b2, acc2);
    acc3 = wmma_bf16(a, b3, acc3);
  }
#pragma unroll
  for (int r = 0; r < 8; ++r) {
    float* orow = out + (m0 + r + 8 * half) * DIM + n0;
    orow[ln]      = acc0[r];
    orow[16 + ln] = acc1[r];
    orow[32 + ln] = acc2[r];
    orow[48 + ln] = acc3[r];
  }
}

// ---------------------------------------------------------------- launcher
extern "C" void kernel_launch(void* const* d_in, const int* in_sizes, int n_in,
                              void* d_out, int out_size, void* d_ws, size_t ws_size,
                              hipStream_t stream) {
  (void)in_sizes; (void)n_in; (void)out_size; (void)ws_size;
  const float* x  = (const float*)d_in[0];
  const int* pos  = (const int*)d_in[1];
  const float* Wq = (const float*)d_in[2];
  const float* Wk = (const float*)d_in[3];
  const float* Wv = (const float*)d_in[4];
  const float* Wo = (const float*)d_in[5];
  float* out = (float*)d_out;

  char* ws = (char*)d_ws;
  size_t o = 0;
  bf16* xb    = (bf16*)(ws + o); o += (size_t)S_LEN * DIM * 2;
  bf16* Wqt   = (bf16*)(ws + o); o += (size_t)DIM * DIM * 2;
  bf16* Wkt   = (bf16*)(ws + o); o += (size_t)DIM * DIM * 2;
  bf16* Wvt   = (bf16*)(ws + o); o += (size_t)DIM * DIM * 2;
  bf16* Wot   = (bf16*)(ws + o); o += (size_t)DIM * DIM * 2;
  bf16* qh    = (bf16*)(ws + o); o += (size_t)S_LEN * DIM * 2;
  bf16* kh    = (bf16*)(ws + o); o += (size_t)S_LEN * DIM * 2;
  bf16* vt    = (bf16*)(ws + o); o += (size_t)S_LEN * DIM * 2;
  bf16* attnb = (bf16*)(ws + o); o += (size_t)S_LEN * DIM * 2;

  convert_x_kernel<<<(S_LEN * DIM + 255) / 256, 256, 0, stream>>>(x, xb, S_LEN * DIM);
  dim3 tb(32, 8), tg(DIM / 32, DIM / 32);
  transpose_convert_kernel<<<tg, tb, 0, stream>>>(Wq, Wqt);
  transpose_convert_kernel<<<tg, tb, 0, stream>>>(Wk, Wkt);
  transpose_convert_kernel<<<tg, tb, 0, stream>>>(Wv, Wvt);
  transpose_convert_kernel<<<tg, tb, 0, stream>>>(Wo, Wot);

  qkv_rope_kernel<<<dim3(S_LEN / 64, NH, 3), 128, 0, stream>>>(xb, Wqt, Wkt, Wvt, pos, qh, kh, vt);
  attn_kernel<<<dim3(S_LEN / 16, NH), 32, 0, stream>>>(qh, kh, vt, attnb);
  oproj_kernel<<<dim3(S_LEN / 64, NH), 128, 0, stream>>>(attnb, Wot, out);
}
